// StereoDPTHead_22574348107859
// MI455X (gfx1250) — compile-verified, matrix operand order checked by reading, not run on hardware
//
#include <hip/hip_runtime.h>
#include <hip/hip_bf16.h>

// ---------------------------------------------------------------------------
// StereoDPT head for MI455X (gfx1250, wave32, WMMA).
//  - Token MLPs / projection: double-buffered GEMM, tiles staged to LDS with
//    global_load_async_to_lds_b128 (ASYNCcnt pipeline), WMMA f32_16x16x32_f16.
//  - 3x3 convs: implicit im2col -> WMMA; weight tiles staged by the Tensor
//    Data Mover (tensor_load_to_lds + s_wait_tensorcnt), A fragments via
//    vectorized ds_load_b128, global_prefetch on upcoming tiles.
//  - GN / resize / cost-volume / elementwise: VALU kernels (bandwidth-trivial).
// ---------------------------------------------------------------------------

typedef __attribute__((ext_vector_type(16))) _Float16 v16h;
typedef __attribute__((ext_vector_type(8)))  _Float16 v8h;
typedef __attribute__((ext_vector_type(8)))  float    v8f;
typedef __attribute__((ext_vector_type(4)))  float    v4f;
typedef __attribute__((ext_vector_type(4)))  int      v4i;
typedef __attribute__((ext_vector_type(4)))  unsigned int u32x4;
typedef __attribute__((ext_vector_type(8)))  int      i32x8;
typedef __attribute__((ext_vector_type(4)))  int      i32x4;

#define TPB 256
#define KC  32

#define FLG_GELU   1
#define FLG_BIAS   2
#define FLG_STORET 4   // store C[n*M + m] (NCHW-style)

#if defined(__HIP_DEVICE_COMPILE__) && __has_builtin(__builtin_amdgcn_global_load_async_to_lds_b128)
#define HAS_ASYNC_LDS 1
#endif
#if defined(__HIP_DEVICE_COMPILE__) && __has_builtin(__builtin_amdgcn_tensor_load_to_lds) && __has_builtin(__builtin_amdgcn_s_wait_tensorcnt)
#define HAS_TDM 1
#endif

typedef __attribute__((address_space(1))) v4i* gv4ip;
typedef __attribute__((address_space(3))) v4i* lv4ip;

__device__ __forceinline__ void async_cp_b128(const float* g, float* l) {
#ifdef HAS_ASYNC_LDS
  __builtin_amdgcn_global_load_async_to_lds_b128((gv4ip)(float*)g, (lv4ip)l, 0, 0);
#else
  *(v4f*)l = *(const v4f*)g;
#endif
}
__device__ __forceinline__ void wait_async_all() {
#ifdef HAS_ASYNC_LDS
#if __has_builtin(__builtin_amdgcn_s_wait_asynccnt)
  __builtin_amdgcn_s_wait_asynccnt(0);
#else
  asm volatile("s_wait_asynccnt 0x0" ::: "memory");
#endif
#endif
}

#ifdef HAS_TDM
// TDM: DMA a 2D tile (tile_h rows x 32 f32) into LDS, padding 4 DWORDs every
// 32 DWORDs -> LDS row stride 36 floats (144B, 16B aligned).  tensor_w/h are
// the remaining extents from the tile origin so OOB reads zero-fill the tail.
__device__ __forceinline__ void tdm_load_tile_f32(const float* gsrc, float* ldst,
                                                  int tile_h, int tensor_w,
                                                  int tensor_h, int row_stride) {
  unsigned lds = (unsigned)(uintptr_t)ldst;
  unsigned long long ga = (unsigned long long)(uintptr_t)gsrc;
  u32x4 g0{};
  g0[0] = 1u;                                    // count=1 (valid user D#)
  g0[1] = lds;                                   // lds_addr
  g0[2] = (unsigned)ga;                          // global_addr[31:0]
  g0[3] = (unsigned)(ga >> 32) | (2u << 30);     // global_addr[56:32] | type=2
  i32x8 g1{};
  g1[0] = (2 << 16) | (1 << 20) | (4 << 22) | (3 << 25); // 4B elems, pad_en, 32dw interval, 4dw pad
  g1[1] = (tensor_w & 0xFFFF) << 16;                         // tensor_dim0 lo
  g1[2] = ((unsigned)tensor_w >> 16) | ((unsigned)(tensor_h & 0xFFFF) << 16);
  g1[3] = ((unsigned)tensor_h >> 16) | (KC << 16);           // tile_dim0 = 32
  g1[4] = tile_h;                                            // tile_dim1
  g1[5] = row_stride;                                        // tensor_dim0_stride
  g1[6] = 0;
  g1[7] = 0;
  i32x4 z4{};
  i32x8 z8{};
  __builtin_amdgcn_tensor_load_to_lds(g0, g1, z4, z4, z8, 0);
}
#endif

__device__ __forceinline__ float gelu_f(float x) {
  return 0.5f * x * (1.0f + erff(x * 0.70710678118654752f));
}
__device__ __forceinline__ int imin(int a, int b) { return a < b ? a : b; }

// ---------------------------------------------------------------------------
// Dense GEMM, exact tiles (M%128==0, N%64==0, K%32==0).
//   C[M,N] = A[M,K] * B  ; BT: B stored [N,K] row-major, else [K,N].
// Block 128x64 (8 waves x 32x32), double-buffered async LDS staging.
// ---------------------------------------------------------------------------
template <bool BT>
__global__ __launch_bounds__(TPB)
void mlp_gemm_wmma(const float* __restrict__ A, const float* __restrict__ Bm,
                   const float* __restrict__ bias, float* __restrict__ Cmat,
                   int M, int N, int K, long long strideA, int flags)
{
  __shared__ float sA[2][128 * 36];   // rows stride 36 floats (144B, 16B-mult)
  __shared__ float sB[2][64 * 36];    // BT: [n][k] stride 36 ; !BT: [k][n] stride 68

  const int tid  = threadIdx.x;
  const int lane = tid & 31;
  const int wid  = tid >> 5;
  const int wm   = (wid >> 1) * 32;
  const int wn   = (wid & 1) * 32;
  const int half = lane >> 4;
  const int l16  = lane & 15;
  const int bm   = blockIdx.x * 128;
  const int bn   = blockIdx.y * 64;
  const int b    = blockIdx.z;
  const float* Ab = A + (long long)b * strideA;

  auto stage = [&](int buf, int k0) {
#pragma unroll
    for (int i = 0; i < 4; ++i) {
      int chunk = i * TPB + tid;
      int mi = chunk >> 3, ci = (chunk & 7) * 4;
      async_cp_b128(Ab + (long long)(bm + mi) * K + k0 + ci, &sA[buf][mi * 36 + ci]);
    }
    if (BT) {
#pragma unroll
      for (int i = 0; i < 2; ++i) {
        int chunk = i * TPB + tid;
        int ni = chunk >> 3, ci = (chunk & 7) * 4;
        async_cp_b128(Bm + (long long)(bn + ni) * K + k0 + ci, &sB[buf][ni * 36 + ci]);
      }
    } else {
      int ki = tid >> 3, ci = (tid & 7) * 4;
      async_cp_b128(Bm + (long long)(k0 + ki) * N + bn + ci, &sB[buf][ki * 68 + ci]);
    }
  };

  v8f acc[2][2] = {};

  stage(0, 0);
  wait_async_all();
  __syncthreads();

  const int nsteps = K / KC;
  for (int s = 0; s < nsteps; ++s) {
    int cur = s & 1, nxt = cur ^ 1;
    if (s + 1 < nsteps) stage(nxt, (s + 1) * KC);

    v16h af[2], bf[2];
#pragma unroll
    for (int i = 0; i < 2; ++i) {
      const float* ap = &sA[cur][(wm + i * 16 + l16) * 36 + 8 * half];
#pragma unroll
      for (int e = 0; e < 8; ++e) { af[i][e] = (_Float16)ap[e]; af[i][8 + e] = (_Float16)ap[16 + e]; }
    }
#pragma unroll
    for (int j = 0; j < 2; ++j) {
      int ncol = wn + j * 16 + l16;
      if (BT) {
        const float* bp = &sB[cur][ncol * 36 + 8 * half];
#pragma unroll
        for (int e = 0; e < 8; ++e) { bf[j][e] = (_Float16)bp[e]; bf[j][8 + e] = (_Float16)bp[16 + e]; }
      } else {
#pragma unroll
        for (int e = 0; e < 16; ++e) {
          int kk = (e < 8 ? e : e + 8) + 8 * half;
          bf[j][e] = (_Float16)sB[cur][kk * 68 + ncol];
        }
      }
    }
#pragma unroll
    for (int i = 0; i < 2; ++i)
#pragma unroll
      for (int j = 0; j < 2; ++j)
        acc[i][j] = __builtin_amdgcn_wmma_f32_16x16x32_f16(
            false, af[i], false, bf[j], (short)0, acc[i][j], false, false);

    wait_async_all();
    __syncthreads();
  }

  float* Cb = Cmat + (long long)b * M * N;
#pragma unroll
  for (int i = 0; i < 2; ++i) {
#pragma unroll
    for (int j = 0; j < 2; ++j) {
      int n = bn + wn + j * 16 + l16;
#pragma unroll
      for (int r = 0; r < 8; ++r) {
        int m = bm + wm + i * 16 + half * 8 + r;
        float v = acc[i][j][r];
        if (flags & FLG_BIAS) v += bias[n];
        if (flags & FLG_GELU) v = gelu_f(v);
        if (flags & FLG_STORET) Cb[(long long)n * M + m] = v;
        else                    Cb[(long long)m * N + n] = v;
      }
    }
  }
}

// ---------------------------------------------------------------------------
// 3x3 conv as implicit-im2col WMMA GEMM. K = Cin*9 (zero-padded to KC by the
// TDM's OOB zero-fill).  Weight tile (64 rows x 32 f32 of the OIHW matrix) is
// DMA'd by the Tensor Data Mover into LDS with 36-float padded row stride.
// Output stored NCHW.  M = H*W (mult of 128), N = Cout (mult of 64).
// ---------------------------------------------------------------------------
__global__ __launch_bounds__(TPB)
void conv3x3_wmma(const float* __restrict__ X, const float* __restrict__ Wgt,
                  float* __restrict__ Cmat, int M, int N, int K,
                  int H, int W, int Cin, long long strideA)
{
  __shared__ _Float16 hA[128 * 40];  // im2col tile, row stride 40 halves (80B)
  __shared__ float    sW[64 * 36];   // weight tile [n][k], row stride 36 f32

  const int tid  = threadIdx.x;
  const int lane = tid & 31;
  const int wid  = tid >> 5;
  const int wm   = (wid >> 1) * 32;
  const int wn   = (wid & 1) * 32;
  const int half = lane >> 4;
  const int l16  = lane & 15;
  const int bm   = blockIdx.x * 128;
  const int bn   = blockIdx.y * 64;
  const int b    = blockIdx.z;
  const float* Xb = X + (long long)b * strideA;

  v8f acc[2][2] = {};

  for (int k0 = 0; k0 < K; k0 += KC) {
#ifdef HAS_TDM
    if (wid == 0)  // one wave issues the weight-tile DMA (EXEC-independent)
      tdm_load_tile_f32(Wgt + (long long)bn * K + k0, sW,
                        64, K - k0, N - bn, K);
#else
    for (int t = tid; t < 64 * KC; t += TPB) {
      int ni = t >> 5, ki = t & 31;
      int k = k0 + ki;
      sW[ni * 36 + ki] = (k < K) ? Wgt[(long long)(bn + ni) * K + k] : 0.0f;
    }
#endif
    if (k0 + KC < K)  // prefetch next weight tile slice into L2
      __builtin_prefetch(&Wgt[(long long)(bn + (tid & 63)) * K + k0 + KC], 0, 1);

    // A tile: on-the-fly im2col with zero halo (f32 -> f16)
    for (int t = tid; t < 128 * KC; t += TPB) {
      int mi = t / KC, ki = t % KC;
      int m = bm + mi, k = k0 + ki;
      float v = 0.0f;
      if (k < K) {
        int ci = k / 9, r = k % 9;
        int y = m / W + (r / 3) - 1;
        int x = m % W + (r % 3) - 1;
        if (y >= 0 && y < H && x >= 0 && x < W)
          v = Xb[(long long)ci * H * W + y * W + x];
      }
      hA[mi * 40 + ki] = (_Float16)v;
    }
#ifdef HAS_TDM
    if (wid == 0) __builtin_amdgcn_s_wait_tensorcnt(0);
#endif
    __syncthreads();

    v16h af[2], bf[2];
#pragma unroll
    for (int i = 0; i < 2; ++i) {
      const _Float16* ap = &hA[(wm + i * 16 + l16) * 40 + 8 * half];
      v8h lo = *(const v8h*)ap;
      v8h hi = *(const v8h*)(ap + 16);
      af[i] = __builtin_shufflevector(lo, hi, 0, 1, 2, 3, 4, 5, 6, 7,
                                      8, 9, 10, 11, 12, 13, 14, 15);
    }
#pragma unroll
    for (int j = 0; j < 2; ++j) {
      const float* bp = &sW[(wn + j * 16 + l16) * 36 + 8 * half];
#pragma unroll
      for (int e = 0; e < 8; ++e) { bf[j][e] = (_Float16)bp[e]; bf[j][8 + e] = (_Float16)bp[16 + e]; }
    }
#pragma unroll
    for (int i = 0; i < 2; ++i)
#pragma unroll
      for (int j = 0; j < 2; ++j)
        acc[i][j] = __builtin_amdgcn_wmma_f32_16x16x32_f16(
            false, af[i], false, bf[j], (short)0, acc[i][j], false, false);
    __syncthreads();
  }

  float* Cb = Cmat + (long long)b * M * N;
#pragma unroll
  for (int i = 0; i < 2; ++i)
#pragma unroll
    for (int j = 0; j < 2; ++j) {
      int n = bn + wn + j * 16 + l16;
#pragma unroll
      for (int r = 0; r < 8; ++r) {
        int m = bm + wm + i * 16 + half * 8 + r;
        Cb[(long long)n * M + m] = acc[i][j][r];   // NCHW
      }
    }
}

// ---------------------------------------------------------------------------
// Elementwise / reduction kernels
// ---------------------------------------------------------------------------
__global__ void tokcat_kernel(const float* __restrict__ tok, float* __restrict__ out,
                              int B_, int T, int D)
{
  int idx = blockIdx.x * TPB + threadIdx.x;
  int total = B_ * T * 2 * D;
  if (idx >= total) return;
  int j = idx % (2 * D);
  int t = (idx / (2 * D)) % T;
  int b = idx / (2 * D * T);
  float v;
  if (j < D) v = tok[((long long)b * (T + 1) + (t + 1)) * D + j];
  else       v = tok[((long long)b * (T + 1)) * D + (j - D)];
  out[idx] = v;
}

__global__ __launch_bounds__(TPB)
void groupnorm_kernel(const float* __restrict__ x, float* __restrict__ out,
                      const float* __restrict__ g, const float* __restrict__ bta,
                      int C, int HW, int gsize, int do_gelu)
{
  int grp = blockIdx.x, b = blockIdx.y;
  long long base = ((long long)b * C + (long long)grp * gsize) * HW;
  int n = gsize * HW;
  float s = 0.0f, s2 = 0.0f;
  for (int i = threadIdx.x; i < n; i += TPB) {
    float v = x[base + i]; s += v; s2 += v * v;
  }
  __shared__ float rs[TPB], rs2[TPB];
  rs[threadIdx.x] = s; rs2[threadIdx.x] = s2;
  __syncthreads();
  for (int o = TPB / 2; o > 0; o >>= 1) {
    if (threadIdx.x < o) { rs[threadIdx.x] += rs[threadIdx.x + o]; rs2[threadIdx.x] += rs2[threadIdx.x + o]; }
    __syncthreads();
  }
  float mean = rs[0] / n;
  float var  = rs2[0] / n - mean * mean;
  float rstd = rsqrtf(var + 1e-5f);
  for (int i = threadIdx.x; i < n; i += TPB) {
    int c = grp * gsize + i / HW;
    float v = (x[base + i] - mean) * rstd * g[c] + bta[c];
    if (do_gelu) v = gelu_f(v);
    out[base + i] = v;
  }
}

__global__ void chan_norm_kernel(const float* __restrict__ in, float* __restrict__ out,
                                 int B_, int C, int HW)
{
  int idx = blockIdx.x * TPB + threadIdx.x;
  if (idx >= B_ * HW) return;
  int b = idx / HW, p = idx % HW;
  long long base = (long long)b * C * HW + p;
  float s = 0.0f;
  for (int c = 0; c < C; ++c) { float v = in[base + (long long)c * HW]; s += v * v; }
  float inv = 1.0f / fmaxf(sqrtf(s), 1e-12f);
  for (int c = 0; c < C; ++c) out[base + (long long)c * HW] = in[base + (long long)c * HW] * inv;
}

__global__ void costvol_kernel(const float* __restrict__ Ln, const float* __restrict__ Rn,
                               float* __restrict__ cv, int B_, int C, int H, int W, int D)
{
  int idx = blockIdx.x * TPB + threadIdx.x;
  int total = B_ * D * H * W;
  if (idx >= total) return;
  int w = idx % W;
  int t = idx / W;
  int h = t % H; t /= H;
  int d = t % D;
  int b = t / D;
  float acc = 0.0f;
  int ws = w - d;
  if (ws >= 0) {
    long long lb = ((long long)b * C) * H * W + h * W + w;
    long long rb = ((long long)b * C) * H * W + h * W + ws;
    int HW = H * W;
    for (int c = 0; c < C; ++c) acc += Ln[lb + (long long)c * HW] * Rn[rb + (long long)c * HW];
  }
  cv[idx] = acc;
}

__global__ void concat_ch_kernel(const float* __restrict__ cv, const float* __restrict__ feat,
                                 float* __restrict__ out, int B_, int D, int Cf, int HW)
{
  int idx = blockIdx.x * TPB + threadIdx.x;
  int Ct = D + Cf;
  int total = B_ * Ct * HW;
  if (idx >= total) return;
  int p = idx % HW;
  int c = (idx / HW) % Ct;
  int b = idx / (HW * Ct);
  float v;
  if (c < D) v = cv[((long long)b * D + c) * HW + p];
  else       v = feat[((long long)b * Cf + (c - D)) * HW + p];
  out[idx] = v;
}

__global__ void bilinear_kernel(const float* __restrict__ in, float* __restrict__ out,
                                int NC, int H, int W, int Ht, int Wt)
{
  int idx = blockIdx.x * TPB + threadIdx.x;
  int total = NC * Ht * Wt;
  if (idx >= total) return;
  int x = idx % Wt;
  int y = (idx / Wt) % Ht;
  int c = idx / (Wt * Ht);
  float fy = ((y + 0.5f) * H) / Ht - 0.5f;
  float fx = ((x + 0.5f) * W) / Wt - 0.5f;
  fy = fminf(fmaxf(fy, 0.0f), (float)(H - 1));
  fx = fminf(fmaxf(fx, 0.0f), (float)(W - 1));
  int y0 = (int)fy, x0 = (int)fx;
  int y1 = imin(y0 + 1, H - 1), x1 = imin(x0 + 1, W - 1);
  float wy = fy - y0, wx = fx - x0;
  const float* p = in + (long long)c * H * W;
  float v00 = p[y0 * W + x0], v01 = p[y0 * W + x1];
  float v10 = p[y1 * W + x0], v11 = p[y1 * W + x1];
  out[idx] = (1 - wy) * ((1 - wx) * v00 + wx * v01) + wy * ((1 - wx) * v10 + wx * v11);
}

__global__ void avgpool_kernel(const float* __restrict__ in, float* __restrict__ out,
                               int NC, int H, int W, int fh, int fw)
{
  int Ht = H / fh, Wt = W / fw;
  int idx = blockIdx.x * TPB + threadIdx.x;
  int total = NC * Ht * Wt;
  if (idx >= total) return;
  int x = idx % Wt;
  int y = (idx / Wt) % Ht;
  int c = idx / (Wt * Ht);
  const float* p = in + (long long)c * H * W;
  float s = 0.0f;
  for (int dy = 0; dy < fh; ++dy)
    for (int dx = 0; dx < fw; ++dx)
      s += p[(y * fh + dy) * W + (x * fw + dx)];
  out[idx] = s / (fh * fw);
}

__global__ void add_kernel(const float* __restrict__ a, const float* __restrict__ b,
                           float* __restrict__ out, int n)
{
  int idx = blockIdx.x * TPB + threadIdx.x;
  if (idx < n) out[idx] = a[idx] + b[idx];
}

__global__ void add_gelu_kernel(const float* __restrict__ a, const float* __restrict__ b,
                                float* __restrict__ out, int n)
{
  int idx = blockIdx.x * TPB + threadIdx.x;
  if (idx < n) out[idx] = gelu_f(a[idx] + b[idx]);
}

__global__ void head1x1_kernel(const float* __restrict__ x, const float* __restrict__ co,
                               const float* __restrict__ bo, float* __restrict__ out,
                               int B_, int C, int HW)
{
  int idx = blockIdx.x * TPB + threadIdx.x;
  if (idx >= B_ * HW) return;
  int b = idx / HW, p = idx % HW;
  long long base = (long long)b * C * HW + p;
  float s = 0.0f;
  for (int c = 0; c < C; ++c) s += co[c] * x[base + (long long)c * HW];
  out[idx] = s + bo[0];
}

// ---------------------------------------------------------------------------
// Host orchestration.  Param index map: 8 token arrays, then params flattened
// jax-style (dict keys sorted at every level), then img_h/img_w.
// ---------------------------------------------------------------------------
#define ENC16_B 8
#define ENC4_B  14
#define ENC8_B  20
#define F16_8_B 26
#define F32_16_B 38
#define F8_4_B  50
#define HEAD_B  62
#define R16L_B  73
#define R16R_B  80
#define R32L_B  87
#define R32R_B  94
#define R4L_B   101
#define R4R_B   108
#define R8L_B   115
#define R8R_B   122
// enc/rcu: b1,b2,c1,c2,g1,g2 ; fuse: rcu1(+0..5) rcu2(+6..11)
// head: b,bo,c,co,g,rcu{..} ; reass: b1,b2,gn_b,gn_g,proj,w1,w2

extern "C" void kernel_launch(void* const* d_in, const int* in_sizes, int n_in,
                              void* d_out, int out_size, void* d_ws, size_t ws_size,
                              hipStream_t stream)
{
  (void)in_sizes; (void)n_in; (void)out_size; (void)ws_size;
  const int B = 2, C = 256, EMB = 768, TOK = 2048;
  auto IN = [&](int i) { return (const float*)d_in[i]; };

  // ---- workspace layout (floats) ----
  float* WS    = (float*)d_ws;
  float* f32L  = WS + 0;
  float* f16L  = WS + 65536;
  float* f16R  = WS + 327680;
  float* f8L   = WS + 589824;
  float* f8R   = WS + 1638400;
  float* f4L   = WS + 2686976;
  float* f4R   = WS + 6881280;
  float* lat16 = WS + 11075584;
  float* lat8  = WS + 11337728;
  float* lat4  = WS + 12386304;
  float* yX    = WS + 16580608;
  float* S     = WS + 20774912;   // stage-aliased scratch
  float* tokcat = S;
  float* h1c    = S + 6291456;
  float* h2c    = S + 9437184;
  float* featG  = S + 12582912;
  float* cvb  = S;
  float* Lnb  = S + 1048576;
  float* Rnb  = S + 5242880;
  float* catb = S + 1048576;
  float* midb = S + 6291456;
  float* upb  = S;
  float* t1b  = S + 4194304;
  float* t2b  = S + 8388608;

  auto ew = [&](int n) { return dim3((n + TPB - 1) / TPB); };

  auto mlp = [&](const float* A, const float* Bm, const float* bias, float* Cp,
                 int M, int N, int K, long long sA, int flags, bool bt) {
    dim3 g(M / 128, N / 64, B);
    if (bt) mlp_gemm_wmma<true><<<g, TPB, 0, stream>>>(A, Bm, bias, Cp, M, N, K, sA, flags);
    else    mlp_gemm_wmma<false><<<g, TPB, 0, stream>>>(A, Bm, bias, Cp, M, N, K, sA, flags);
  };
  auto conv = [&](const float* X, const float* Wg, float* Cp, int H, int Wd, int Cin) {
    int M = H * Wd;
    dim3 g(M / 128, C / 64, B);
    conv3x3_wmma<<<g, TPB, 0, stream>>>(X, Wg, Cp, M, C, Cin * 9, H, Wd, Cin,
                                        (long long)Cin * H * Wd);
  };
  auto gnorm = [&](const float* x, float* o, const float* g, const float* bt,
                   int HW, int do_gelu) {
    groupnorm_kernel<<<dim3(32, B), TPB, 0, stream>>>(x, o, g, bt, C, HW, C / 32, do_gelu);
  };

  // ---- reassemble: tokens -> feature map at target resolution ----
  auto reassemble = [&](const float* tok, int pb, float* target, int mode) {
    const float* b1   = IN(pb + 0);
    const float* b2   = IN(pb + 1);
    const float* gn_b = IN(pb + 2);
    const float* gn_g = IN(pb + 3);
    const float* proj = IN(pb + 4);
    const float* w1   = IN(pb + 5);
    const float* w2   = IN(pb + 6);
    tokcat_kernel<<<ew(B * TOK * 2 * EMB), TPB, 0, stream>>>(tok, tokcat, B, TOK, EMB);
    mlp(tokcat, w1, b1, h1c, TOK, EMB, 2 * EMB, (long long)TOK * 2 * EMB,
        FLG_BIAS | FLG_GELU, false);
    mlp(h1c, w2, b2, h2c, TOK, EMB, EMB, (long long)TOK * EMB, FLG_BIAS, false);
    float* fdst = (mode == 0) ? target : featG;
    mlp(h2c, proj, nullptr, fdst, TOK, C, EMB, (long long)TOK * EMB, FLG_STORET, true);
    gnorm(fdst, fdst, gn_g, gn_b, TOK, 1);
    int NC = B * C;
    if (mode == 1) avgpool_kernel<<<ew(NC * 8 * 16), TPB, 0, stream>>>(featG, target, NC, 32, 64, 4, 4);
    else if (mode == 2) avgpool_kernel<<<ew(NC * 16 * 32), TPB, 0, stream>>>(featG, target, NC, 32, 64, 2, 2);
    else if (mode == 3) bilinear_kernel<<<ew(NC * 64 * 128), TPB, 0, stream>>>(featG, target, NC, 32, 64, 64, 128);
  };

  reassemble(IN(3), R32L_B, f32L, 1);  // t32L -> 8x16
  reassemble(IN(2), R16L_B, f16L, 2);  // t16L -> 16x32
  reassemble(IN(6), R16R_B, f16R, 2);
  reassemble(IN(1), R8L_B,  f8L,  0);  // t8L -> 32x64
  reassemble(IN(5), R8R_B,  f8R,  0);
  reassemble(IN(0), R4L_B,  f4L,  3);  // t4L -> 64x128
  reassemble(IN(4), R4R_B,  f4R,  3);

  // ---- cost volume + encoder per scale ----
  auto do_scale = [&](const float* fL, const float* fR, int H, int W, int D,
                      int eb, float* lat) {
    int HW = H * W;
    chan_norm_kernel<<<ew(B * HW), TPB, 0, stream>>>(fL, Lnb, B, C, HW);
    chan_norm_kernel<<<ew(B * HW), TPB, 0, stream>>>(fR, Rnb, B, C, HW);
    costvol_kernel<<<ew(B * D * HW), TPB, 0, stream>>>(Lnb, Rnb, cvb, B, C, H, W, D);
    int Cin = D + C;
    concat_ch_kernel<<<ew(B * Cin * HW), TPB, 0, stream>>>(cvb, fL, catb, B, D, C, HW);
    conv(catb, IN(eb + 2), midb, H, W, Cin);
    gnorm(midb, midb, IN(eb + 4), IN(eb + 0), HW, 1);
    conv(midb, IN(eb + 3), lat, H, W, C);
    gnorm(lat, lat, IN(eb + 5), IN(eb + 1), HW, 1);
  };
  do_scale(f16L, f16R, 16, 32, 13, ENC16_B, lat16);
  do_scale(f8L,  f8R,  32, 64, 25, ENC8_B,  lat8);
  do_scale(f4L,  f4R,  64, 128, 49, ENC4_B, lat4);

  // ---- residual conv unit (in-place on x) ----
  auto rcu = [&](float* x, int rb, int H, int W) {
    int HW = H * W;
    conv(x, IN(rb + 2), t1b, H, W, C);
    gnorm(t1b, t1b, IN(rb + 4), IN(rb + 0), HW, 1);
    conv(t1b, IN(rb + 3), t2b, H, W, C);
    gnorm(t2b, t2b, IN(rb + 5), IN(rb + 1), HW, 0);
    add_gelu_kernel<<<ew(B * C * HW), TPB, 0, stream>>>(t2b, x, x, B * C * HW);
  };

  // ---- fuse: yX = rcu(rcu(bilinear(src) + lat)) ----
  auto fuse = [&](const float* src, const float* lat, int Hs, int Ws, int Ht, int Wt, int fb) {
    int NC = B * C;
    bilinear_kernel<<<ew(NC * Ht * Wt), TPB, 0, stream>>>(src, upb, NC, Hs, Ws, Ht, Wt);
    add_kernel<<<ew(NC * Ht * Wt), TPB, 0, stream>>>(upb, lat, yX, NC * Ht * Wt);
    rcu(yX, fb + 0, Ht, Wt);
    rcu(yX, fb + 6, Ht, Wt);
  };
  fuse(f32L, lat16, 8, 16, 16, 32, F32_16_B);
  fuse(yX,   lat8, 16, 32, 32, 64, F16_8_B);
  fuse(yX,   lat4, 32, 64, 64, 128, F8_4_B);

  // ---- head ----
  rcu(yX, HEAD_B + 5, 64, 128);
  int HW = 64 * 128;
  conv(yX, IN(HEAD_B + 2), t1b, 64, 128, C);
  gnorm(t1b, t1b, IN(HEAD_B + 4), IN(HEAD_B + 0), HW, 1);
  head1x1_kernel<<<ew(B * HW), TPB, 0, stream>>>(t1b, IN(HEAD_B + 3), IN(HEAD_B + 1),
                                                 (float*)d_out, B, C, HW);
}
